// LSTM5D_32899449487586
// MI455X (gfx1250) — matrix operand order repaired and checked by reading, not compiled, not run on hardware
//
#include <hip/hip_runtime.h>

// ---------------------------------------------------------------------------
// 2-layer pixelwise ConvLSTM (1x1 conv) for gfx1250 using v_wmma_f32_16x16x32_f16
//   x:  [B=8, T=32, Cin=64, H=32, W=32] f32
//   W0: [512, 192] f32   b0: [512]
//   W1: [512, 256] f32   b1: [512]
//   out: y1 [8,32,128,32,32] ++ hT [8,128,32,32] ++ cT [8,128,32,32] (f32)
//
// Per step: GEMM  M=8192 (B*H*W), N=512 (4 gates x HID), K=Cin+HID.
// Block = 256 threads (8 waves) handles 32 rows x all 512 gates.
// Wave w owns hid block [16w,16w+16); per K-step it holds the 4 gate B-fragments
// in registers and reuses them across TWO 16-row M-subtiles (8 wmma per B fetch),
// halving W L2 traffic vs a 16-row block. LSTM cell update is register-local
// because the i,f,o,g D-fragments share (row, hid) <-> (vgpr, lane) mapping.
// ---------------------------------------------------------------------------

typedef __attribute__((ext_vector_type(16))) _Float16 v16h;
typedef __attribute__((ext_vector_type(8)))  _Float16 v8h;
typedef __attribute__((ext_vector_type(8)))  float    v8f;

#define T_STEPS 32
#define HID     128
#define SPAT    1024            // H*W
#define NBATCH  8
#define NROW    (NBATCH * SPAT) // 8192 spatial rows per step
#define MROWS   32              // rows per block (2 M-subtiles of 16)

__device__ __forceinline__ float sigf(float x)     { return 1.0f / (1.0f + __expf(-x)); }
__device__ __forceinline__ float tanhfast(float x) { return 2.0f / (1.0f + __expf(-2.0f * x)) - 1.0f; }

template <int CIN>
__global__ __launch_bounds__(256)
void lstm_step_kernel(const float*    __restrict__ x32,
                      const _Float16* __restrict__ x16,
                      const _Float16* __restrict__ Wh,    // [512][K] f16 row-major
                      const float*    __restrict__ bias,  // [512]
                      _Float16*       __restrict__ hbuf,  // [B][HID][S] f16 (in/out)
                      float*          __restrict__ cbuf,  // [B*S][HID] f32 (in/out)
                      float*          __restrict__ yout,  // layer1: y output base or null
                      float*          __restrict__ hTout, // final step or null
                      float*          __restrict__ cTout, // final step or null
                      int t)
{
    constexpr int K    = CIN + HID;   // 192 or 256
    constexpr int KPAD = K + 8;       // 16B-aligned row stride, bank stagger

    __shared__ alignas(16) _Float16 zlds[MROWS][KPAD];      // A tile (32 rows x K) f16
    __shared__ alignas(16) _Float16 hstage[MROWS][HID + 8]; // h_new f16 staging
    __shared__ alignas(16) float    ystage[MROWS][HID + 8]; // h_new f32 staging
    __shared__ alignas(16) float    cstage[MROWS][HID + 8]; // c_new f32 staging

    const int tid   = threadIdx.x;
    const int m0    = blockIdx.x * MROWS;   // first global spatial row of this block
    const int b     = m0 >> 10;             // batch index (SPAT = 1024 rows per batch)
    const int sbase = m0 & (SPAT - 1);
    const int sl    = tid & 31;             // local row for cooperative load/store

    // ---- stage z = concat(x_t, h_prev) into LDS as f16 (32-lane coalesced) ----
    for (int c = tid >> 5; c < K; c += 8) {
        float v;
        if (c < CIN) {
            if constexpr (CIN == 64)
                v = x32[(((size_t)b * T_STEPS + t) * CIN + c) * SPAT + sbase + sl];
            else
                v = (float)x16[((size_t)b * CIN + c) * SPAT + sbase + sl];
        } else {
            v = (float)hbuf[((size_t)b * HID + (c - CIN)) * SPAT + sbase + sl];
        }
        zlds[sl][c] = (_Float16)v;
    }
    __syncthreads();

    const int lane = tid & 31;
    const int ln16 = lane & 15;
    const int hi   = lane >> 4;         // lane half selects K sub-block per ISA layout
    const int wave = tid >> 5;          // 0..7 -> hid block
    const int hidc = wave * 16 + ln16;  // hid column 0..127 (also D-fragment column)

    // Accumulators: [gate][m-subtile]; start at 0, bias added in elementwise phase
    v8f acc[4][2];
    #pragma unroll
    for (int g = 0; g < 4; ++g)
        #pragma unroll
        for (int mt = 0; mt < 2; ++mt)
            acc[g][mt] = (v8f){0.f, 0.f, 0.f, 0.f, 0.f, 0.f, 0.f, 0.f};

    // ---- GEMM: per K-step, fetch 4 B fragments once, reuse over 2 M-subtiles ----
    #pragma unroll
    for (int kb = 0; kb < K / 32; ++kb) {
        // B fragments (32x16 f16): lane = column n, contiguous 16-K run per lane half
        const int kB = kb * 32 + hi * 16;
        v16h bf[4];
        #pragma unroll
        for (int g = 0; g < 4; ++g)
            bf[g] = *(const v16h*)(Wh + (size_t)(g * HID + hidc) * K + kB);

        const int kA = kb * 32 + hi * 16; // A: contiguous 16-K run per lane half
        #pragma unroll
        for (int mt = 0; mt < 2; ++mt) {
            const int arow = mt * 16 + ln16;
            v8h a0 = *(const v8h*)&zlds[arow][kA];
            v8h a1 = *(const v8h*)&zlds[arow][kA + 8];
            v16h af = __builtin_shufflevector(a0, a1,
                       0, 1, 2, 3, 4, 5, 6, 7, 8, 9, 10, 11, 12, 13, 14, 15);
            #pragma unroll
            for (int g = 0; g < 4; ++g)
                acc[g][mt] = __builtin_amdgcn_wmma_f32_16x16x32_f16(
                                 false, af, false, bf[g], (short)0, acc[g][mt],
                                 false, false);
        }
    }

    // ---- LSTM cell update, register-local per (m, hid) ----
    const float bi = bias[0 * HID + hidc];
    const float bf_ = bias[1 * HID + hidc];
    const float bo = bias[2 * HID + hidc];
    const float bg = bias[3 * HID + hidc];

    #pragma unroll
    for (int mt = 0; mt < 2; ++mt) {
        #pragma unroll
        for (int r = 0; r < 8; ++r) {
            const int ml = mt * 16 + hi * 8 + r;                 // local row (D layout)
            const size_t cidx = (size_t)(m0 + ml) * HID + hidc;  // coalesced over hid
            const float cp = cbuf[cidx];
            const float iv = sigf(acc[0][mt][r] + bi);
            const float fv = sigf(acc[1][mt][r] + bf_);
            const float ov = sigf(acc[2][mt][r] + bo);
            const float gv = tanhfast(acc[3][mt][r] + bg);
            const float cn = fv * cp + iv * gv;
            const float hn = ov * tanhfast(cn);
            cbuf[cidx]       = cn;
            hstage[ml][hidc] = (_Float16)hn;
            ystage[ml][hidc] = hn;
            cstage[ml][hidc] = cn;
        }
    }
    __syncthreads();

    // ---- coalesced channel-major global writes (32-lane rows) ----
    for (int c = tid >> 5; c < HID; c += 8) {
        const size_t bo2 = ((size_t)b * HID + c) * SPAT + sbase + sl;
        hbuf[bo2] = hstage[sl][c];
        if (yout)  yout[(((size_t)b * T_STEPS + t) * HID + c) * SPAT + sbase + sl] = ystage[sl][c];
        if (hTout) hTout[bo2] = ystage[sl][c];
        if (cTout) cTout[bo2] = cstage[sl][c];
    }
}

__global__ void convert_w_f16(const float* __restrict__ src, _Float16* __restrict__ dst, int n)
{
    const int i = blockIdx.x * blockDim.x + threadIdx.x;
    if (i < n) dst[i] = (_Float16)src[i];
}

__global__ void zero_bytes16(uint4* __restrict__ p, int n16)
{
    const int i = blockIdx.x * blockDim.x + threadIdx.x;
    if (i < n16) p[i] = make_uint4(0u, 0u, 0u, 0u);
}

extern "C" void kernel_launch(void* const* d_in, const int* in_sizes, int n_in,
                              void* d_out, int out_size, void* d_ws, size_t ws_size,
                              hipStream_t stream)
{
    (void)in_sizes; (void)n_in; (void)out_size; (void)ws_size;

    const float* x  = (const float*)d_in[0];
    const float* W0 = (const float*)d_in[1];
    const float* b0 = (const float*)d_in[2];
    const float* W1 = (const float*)d_in[3];
    const float* b1 = (const float*)d_in[4];

    // Workspace layout (all offsets 256B aligned):
    //   [0       ) W0h f16 512*192 = 196608 B
    //   [196608  ) W1h f16 512*256 = 262144 B
    //   [458752  ) h0  f16 [8][128][1024] = 2 MiB
    //   [2555904 ) h1  f16 [8][128][1024] = 2 MiB
    //   [4653056 ) c0  f32 [8192][128]    = 4 MiB
    //   [8847360 ) c1  f32 [8192][128]    = 4 MiB   (end 13041664)
    char* ws = (char*)d_ws;
    _Float16* W0h = (_Float16*)(ws + 0);
    _Float16* W1h = (_Float16*)(ws + 196608);
    _Float16* h0  = (_Float16*)(ws + 458752);
    _Float16* h1  = (_Float16*)(ws + 2555904);
    float*    c0  = (float*)   (ws + 4653056);
    float*    c1  = (float*)   (ws + 8847360);

    float* y  = (float*)d_out;
    float* hT = y  + (size_t)NBATCH * T_STEPS * HID * SPAT;  // 33,554,432
    float* cT = hT + (size_t)NBATCH * HID * SPAT;            // +1,048,576

    // Weights -> f16 once per call; zero recurrent state (deterministic per call).
    convert_w_f16<<<(512 * 192 + 255) / 256, 256, 0, stream>>>(W0, W0h, 512 * 192);
    convert_w_f16<<<(512 * 256 + 255) / 256, 256, 0, stream>>>(W1, W1h, 512 * 256);
    zero_bytes16<<<786432 / 256, 256, 0, stream>>>((uint4*)(ws + 458752), 786432);

    for (int t = 0; t < T_STEPS; ++t) {
        lstm_step_kernel<64><<<NROW / MROWS, 256, 0, stream>>>(
            x, nullptr, W0h, b0, h0, c0, nullptr, nullptr, nullptr, t);
        const bool last = (t == T_STEPS - 1);
        lstm_step_kernel<128><<<NROW / MROWS, 256, 0, stream>>>(
            nullptr, h0, W1h, b1, h1, c1, y,
            last ? hT : nullptr, last ? cT : nullptr, t);
    }
}
